// TopKRouter_43490838839444
// MI455X (gfx1250) — matrix operand order repaired and checked by reading, not compiled, not run on hardware
//
#include <hip/hip_runtime.h>

// ---------------------------------------------------------------------------
// MoE top-2 router for MI455X (gfx1250, wave32).
// logits = x[16384,4096] @ gate_w[64,4096]^T via v_wmma_f32_16x16x32_bf16,
// with TDM (tensor_load_to_lds) double-buffered staging of f32 tiles,
// then mask -> softmax -> top-2 -> renormalize, fused in one kernel.
// ---------------------------------------------------------------------------

typedef __attribute__((ext_vector_type(16))) __bf16       v16bf;
typedef __attribute__((ext_vector_type(8)))  __bf16       v8bf;
typedef __attribute__((ext_vector_type(4)))  __bf16       v4bf;
typedef __attribute__((ext_vector_type(8)))  float        v8f;
typedef __attribute__((ext_vector_type(4)))  unsigned int v4u;
typedef __attribute__((ext_vector_type(8)))  int          v8i;
typedef __attribute__((ext_vector_type(4)))  int          v4i;

#define TOKENS 16384
#define HIDDEN 4096
#define NEXP   64
#define TOPK   2

#define TTILE  64      // tokens per workgroup
#define KC     128     // K chunk
#define NCHUNK (HIDDEN / KC)
#define SSTR   132     // f32 staging row stride (128 + TDM pad of 4 dwords)
#define LDA    136     // bf16 compute row stride (128 + 8 pad elems)
#define LDL    68      // logits row stride (64 + 4 pad f32)

union AFrag { v16bf v; v8bf h[2]; };

__device__ __forceinline__ unsigned lds_off(const void* p) {
    // generic pointer to LDS: low 32 bits are the LDS byte offset
    return (unsigned)(unsigned long)(uintptr_t)p;
}

// Issue one TDM 2-D tile load: tile_dim = KC x 64 rows of f32, row stride
// `stride0` elems, destination LDS rows padded to SSTR f32 via pad fields.
__device__ __forceinline__ void tdm_load_tile_f32(const float* g, unsigned dst,
                                                  unsigned tensor_d0,
                                                  unsigned tensor_d1,
                                                  unsigned long stride0) {
    const unsigned long ga = (unsigned long)(uintptr_t)g;
    v4u g0;
    g0[0] = 1u;                                       // count=1, user D#
    g0[1] = dst;                                      // lds_addr (bytes)
    g0[2] = (unsigned)(ga & 0xFFFFFFFFu);             // global_addr[31:0]
    g0[3] = (unsigned)((ga >> 32) & 0x01FFFFFFu)      // global_addr[56:32]
          | (2u << 30);                               // type = 2 (image)
    v8i g1;
    g1[0] = (int)((2u << 16)                          // data_size = 4 bytes
          | (1u << 20)                                // pad_enable
          | (6u << 22)                                // pad_interval: 128 dwords
          | (3u << 25));                              // pad_amount: 4 dwords
    g1[1] = (int)((tensor_d0 & 0xFFFFu) << 16);       // tensor_dim0[15:0]
    g1[2] = (int)((tensor_d0 >> 16)                   // tensor_dim0[31:16]
          | ((tensor_d1 & 0xFFFFu) << 16));           // tensor_dim1[15:0]
    g1[3] = (int)((tensor_d1 >> 16)                   // tensor_dim1[31:16]
          | ((unsigned)KC << 16));                    // tile_dim0 = 128
    g1[4] = (int)(unsigned)TTILE;                     // tile_dim1 = 64, tile_dim2 = 0
    g1[5] = (int)(stride0 & 0xFFFFFFFFu);             // tensor_dim0_stride[31:0]
    g1[6] = (int)((stride0 >> 32) & 0xFFFFu);         // stride[47:32], dim1_stride=0
    g1[7] = 0;
    v4i g2 = {0, 0, 0, 0};                            // 2-D tensor: dims 2/3 unused
    v4i g3 = {0, 0, 0, 0};
    v8i g4 = {0, 0, 0, 0, 0, 0, 0, 0};                // unused trailing group
    __builtin_amdgcn_tensor_load_to_lds(g0, g1, g2, g3, g4, 0);
}

__global__ __launch_bounds__(256)
void moe_topk_router_kernel(const float* __restrict__ x,
                            const unsigned char* __restrict__ expert_mask,
                            const float* __restrict__ gate_w,
                            float* __restrict__ out_logits,
                            float* __restrict__ out_probs,
                            float* __restrict__ out_topk_p,
                            float* __restrict__ out_topk_i)
{
    __shared__ float  Sxa[2][TTILE * SSTR];   // TDM staging: x tile (f32)
    __shared__ float  Sxb[2][NEXP  * SSTR];   // TDM staging: gate_w tile (f32)
    __shared__ __bf16 As[TTILE * LDA];        // bf16 compute tile (x)
    __shared__ __bf16 Bs[NEXP  * LDA];        // bf16 compute tile (gate_w)
    __shared__ float  Ls[TTILE * LDL];        // logits / probs scratch

    const int tid  = threadIdx.x;
    const int lane = tid & 31;
    const int wave = tid >> 5;
    const int tokenBase = blockIdx.x * TTILE;

    // 16 (4x4) 16x16 output tiles, 8 waves -> 2 tiles per wave
    const int rowTile = wave >> 1;
    const int colPair = (wave & 1) * 2;
    const int half    = (lane >> 4) & 1;
    const int mrow    = rowTile * 16 + (lane & 15);
    const int e0      = colPair * 16 + (lane & 15);
    const int e1      = e0 + 16;

    const unsigned offA[2] = { lds_off(&Sxa[0][0]), lds_off(&Sxa[1][0]) };
    const unsigned offB[2] = { lds_off(&Sxb[0][0]), lds_off(&Sxb[1][0]) };

    v8f acc0 = {};
    v8f acc1 = {};

    // Prologue: wave 0 kicks off TDM for chunk 0 into staging buffer 0.
    if (wave == 0) {
        tdm_load_tile_f32(x + (long)tokenBase * HIDDEN, offA[0], HIDDEN, TOKENS, HIDDEN);
        tdm_load_tile_f32(gate_w,                       offB[0], HIDDEN, NEXP,   HIDDEN);
    }

    for (int c = 0; c < NCHUNK; ++c) {
        const int cur = c & 1;
        const int nxt = cur ^ 1;

        if (wave == 0) __builtin_amdgcn_s_wait_tensorcnt(0);  // chunk c landed
        __syncthreads();   // broadcast; also frees stage[nxt] + bf16 bufs

        // Overlap: issue TDM for chunk c+1 while we convert + compute chunk c.
        if (wave == 0 && c + 1 < NCHUNK) {
            const int kn = (c + 1) * KC;
            tdm_load_tile_f32(x + (long)tokenBase * HIDDEN + kn, offA[nxt],
                              HIDDEN, TOKENS, HIDDEN);
            tdm_load_tile_f32(gate_w + kn, offB[nxt], HIDDEN, NEXP, HIDDEN);
        }

        // ---- LDS f32 staging -> LDS bf16 compute tiles ----
        // one row per wave per step; b128 reads / b64 writes, conflict-free
#pragma unroll
        for (int i = 0; i < 8; ++i) {
            const int r = wave + 8 * i;
            const float4 va = *(const float4*)&Sxa[cur][r * SSTR + lane * 4];
            const float4 vb = *(const float4*)&Sxb[cur][r * SSTR + lane * 4];
            v4bf a4 = { (__bf16)va.x, (__bf16)va.y, (__bf16)va.z, (__bf16)va.w };
            v4bf b4 = { (__bf16)vb.x, (__bf16)vb.y, (__bf16)vb.z, (__bf16)vb.w };
            *(v4bf*)(&As[r * LDA + lane * 4]) = a4;
            *(v4bf*)(&Bs[r * LDA + lane * 4]) = b4;
        }
        __syncthreads();   // bf16 tiles ready

        // ---- 4 WMMA K-steps of 32 over this chunk ----
#pragma unroll
        for (int ks = 0; ks < KC; ks += 32) {
            AFrag a, b0, b1;
            // A 16x32 bf16, contiguous k order chosen here (same as B): lane<16 -> K[0..15]; lane>=16 -> K[16..31]
            const __bf16* ap = &As[mrow * LDA + ks + half * 16];
            a.h[0] = *(const v8bf*)(ap);
            a.h[1] = *(const v8bf*)(ap + 8);
            // B 32x16 bf16, same contiguous k order as A: lane<16 -> K[0..15]; lane>=16 -> K[16..31]
            const __bf16* bp0 = &Bs[e0 * LDA + ks + half * 16];
            b0.h[0] = *(const v8bf*)(bp0);
            b0.h[1] = *(const v8bf*)(bp0 + 8);
            const __bf16* bp1 = &Bs[e1 * LDA + ks + half * 16];
            b1.h[0] = *(const v8bf*)(bp1);
            b1.h[1] = *(const v8bf*)(bp1 + 8);

            acc0 = __builtin_amdgcn_wmma_f32_16x16x32_bf16(
                       false, a.v, false, b0.v, (short)0, acc0, false, false);
            acc1 = __builtin_amdgcn_wmma_f32_16x16x32_bf16(
                       false, a.v, false, b1.v, (short)0, acc1, false, false);
        }
    }

    // ---- scatter accumulators to LDS (C layout: VGPR i -> M=i or i+8) ----
#pragma unroll
    for (int i = 0; i < 8; ++i) {
        const int r = rowTile * 16 + half * 8 + i;
        Ls[r * LDL + e0] = acc0[i];
        Ls[r * LDL + e1] = acc1[i];
    }
    __syncthreads();

    // ---- apply expert mask, emit logits (coalesced) ----
    const int  lrow  = tid >> 2;
    const int  cbase = (tid & 3) * 16;
    const long grow  = (long)(tokenBase + lrow) * NEXP;
#pragma unroll
    for (int j = 0; j < 16; ++j) {
        const int c = cbase + j;
        float l = Ls[lrow * LDL + c];
        if (!expert_mask[c]) l = -__builtin_inff();
        Ls[lrow * LDL + c]   = l;
        out_logits[grow + c] = l;
    }
    __syncthreads();

    // ---- per-token softmax + top-2 (one thread per token) ----
    if (tid < TTILE) {
        const int t = tid;
        float m   = -__builtin_inff();
        float b1v = -__builtin_inff(); int b1i = 0;
        float b2v = -__builtin_inff(); int b2i = 0;
#pragma unroll 4
        for (int e = 0; e < NEXP; ++e) {
            const float l = Ls[t * LDL + e];
            m = fmaxf(m, l);
            if (l > b1v)      { b2v = b1v; b2i = b1i; b1v = l; b1i = e; }
            else if (l > b2v) { b2v = l;   b2i = e; }
        }
        float s = 0.0f;
#pragma unroll 4
        for (int e = 0; e < NEXP; ++e) {
            const float p = __expf(Ls[t * LDL + e] - m);
            s += p;
            Ls[t * LDL + e] = p;
        }
        const float inv = 1.0f / s;
#pragma unroll 4
        for (int e = 0; e < NEXP; ++e)
            Ls[t * LDL + e] *= inv;

        const float p1 = __expf(b1v - m) * inv;
        const float p2 = __expf(b2v - m) * inv;
        const float rn = 1.0f / (p1 + p2);
        const long  tg = (long)(tokenBase + t) * TOPK;
        out_topk_p[tg + 0] = p1 * rn;
        out_topk_p[tg + 1] = p2 * rn;
        out_topk_i[tg + 0] = (float)b1i;
        out_topk_i[tg + 1] = (float)b2i;
    }
    __syncthreads();

    // ---- emit probs (coalesced) ----
#pragma unroll
    for (int j = 0; j < 16; ++j)
        out_probs[grow + cbase + j] = Ls[lrow * LDL + cbase + j];
}

extern "C" void kernel_launch(void* const* d_in, const int* in_sizes, int n_in,
                              void* d_out, int out_size, void* d_ws, size_t ws_size,
                              hipStream_t stream) {
    const float*         x    = (const float*)d_in[0];
    const unsigned char* mask = (const unsigned char*)d_in[1];   // jnp bool -> 1 byte
    const float*         gw   = (const float*)d_in[2];

    float* out    = (float*)d_out;
    float* logits = out;                                  // [TOKENS, 64]
    float* probs  = logits + (long)TOKENS * NEXP;         // [TOKENS, 64]
    float* tkp    = probs  + (long)TOKENS * NEXP;         // [TOKENS, 2]
    float* tki    = tkp    + (long)TOKENS * TOPK;         // [TOKENS, 2] (as float)

    dim3 grid(TOKENS / TTILE);   // 256 workgroups
    dim3 block(256);             // 8 wave32 waves
    moe_topk_router_kernel<<<grid, block, 0, stream>>>(x, mask, gw,
                                                       logits, probs, tkp, tki);
}
